// PolynomialRegression_75385265979710
// MI455X (gfx1250) — matrix operand order repaired and checked by reading, not compiled, not run on hardware
//
#include <hip/hip_runtime.h>
#include <hip/hip_bf16.h>

typedef __attribute__((ext_vector_type(16))) _Float16 v16h;
typedef __attribute__((ext_vector_type(8)))  _Float16 v8h;
typedef __attribute__((ext_vector_type(8)))  float    v8f;

#define BATCH 4096
#define DIN   256
#define NOUT  10
#define KPAD  288                 // 256 + 1 (constant coord) padded to 9*32
#define NTILES 17                 // cover poly-columns 0..271 (col 256 = bias+linear)
#define KTILES 9
#define NUMFEAT 33153             // 1 + 256 + 256*257/2
#define UNITS (NOUT * NTILES)     // 170
#define TILEH (16 * KPAD)         // halves per 16-row X̂ tile

// ---------------------------------------------------------------------------
// Prep 1: X̂ (B x 288) in f16.  x̂[b, j<256] = x[b,j]; x̂[b,256] = 1; rest 0.
// ---------------------------------------------------------------------------
__global__ void build_xhat(const float* __restrict__ x, _Float16* __restrict__ xh) {
    int idx = blockIdx.x * blockDim.x + threadIdx.x;
    if (idx >= BATCH * KPAD) return;
    int b = idx / KPAD, j = idx - b * KPAD;
    float v = (j < DIN) ? x[b * DIN + j] : ((j == DIN) ? 1.0f : 0.0f);
    xh[idx] = (_Float16)v;
}

// ---------------------------------------------------------------------------
// Prep 2: A-matrix fragments of M̂ᵀ_o, pre-shuffled into the exact per-lane
// 16-bit WMMA *A* 16x32 layout (ISA 7.12.2):
//   lane L: row m = L%16 (poly col c = nt*16 + m), khalf = L/16
//   element e (v=e>>1,t=e&1): Koff = (v<4?0:16) + (v&3)*2 + khalf*8 + t
// A[m, K] = M̂ᵀ[c, K] = M̂[K, c] with
// M̂[i,j] = W_quad[tri(i,j)] (i<=j<256) | W_lin[i] (j==256) | W[o,0] (i==j==256) | 0
// Frag index: (((o*17+nt)*9 + kt)*32 + lane)*16 + e
// ---------------------------------------------------------------------------
__global__ void build_mfrag(const float* __restrict__ W, _Float16* __restrict__ mf) {
    int t = blockIdx.x * blockDim.x + threadIdx.x;
    if (t >= UNITS * KTILES * 32) return;
    int lane = t & 31;
    int r = t >> 5;
    int kt = r % KTILES; r /= KTILES;
    int nt = r % NTILES;
    int o  = r / NTILES;

    int c     = (lane & 15) + nt * 16;   // poly column of M̂ (= A row)
    int khalf = lane >> 4;
    const float* Wo = W + (size_t)o * NUMFEAT;

    _Float16 vals[16];
#pragma unroll
    for (int e = 0; e < 16; ++e) {
        int v = e >> 1, tb = e & 1;
        int K = kt * 32 + ((v < 4) ? 0 : 16) + (v & 3) * 2 + khalf * 8 + tb;
        float val = 0.0f;
        if (K < DIN && c < DIN) {
            if (K <= c) {
                int tri = K * DIN - (K * (K - 1)) / 2 + (c - K);
                val = Wo[1 + DIN + tri];
            }
        } else if (K < DIN && c == DIN) {
            val = Wo[1 + K];               // linear term column
        } else if (K == DIN && c == DIN) {
            val = Wo[0];                   // bias
        }
        vals[e] = (_Float16)val;
    }
    uint4* dst = (uint4*)(mf + ((size_t)t << 4));
    dst[0] = ((const uint4*)vals)[0];
    dst[1] = ((const uint4*)vals)[1];
}

// ---------------------------------------------------------------------------
// Main: block = TWO 16-row batch tiles (32 rows); 8 waves sweep 170
// (o, n_tile) units. Each A fragment streamed from L2 feeds 2 WMMAs (one per
// tile), doubling arithmetic intensity on the A stream (384 MB -> 192 MB of
// L2 traffic) and giving two independent accumulator chains per unit.
// D = M̂ᵀ_tile x X̂ᵀ_tile: batch row = lane, reduce dim = VGPR index, so the
// tail per tile is 1 LDS b128 load + 8 FMA + 1 ds_add_f32.
// ---------------------------------------------------------------------------
__global__ __launch_bounds__(256) void poly_wmma(const _Float16* __restrict__ xh,
                                                 const _Float16* __restrict__ mf,
                                                 float* __restrict__ out) {
    __shared__ _Float16 xt[2 * TILEH];   // 18432 B (two 16x288 tiles)
    __shared__ float    q[32 * NOUT];    // 1280 B

    const int tid = threadIdx.x;
    const int bt  = blockIdx.x;          // 32-row batch tile

    // cooperative load of both X̂ tiles: 2*16*288 halves = 4608 dwords
    {
        const uint32_t* src = (const uint32_t*)(xh + (size_t)bt * 2 * TILEH);
        uint32_t* dst = (uint32_t*)xt;
        for (int i = tid; i < TILEH; i += 256) dst[i] = src[i];   // TILEH dwords = 2 tiles
    }
    if (tid < 32 * NOUT) q[tid] = 0.0f;
    __syncthreads();

    const int wave  = tid >> 5;
    const int lane  = tid & 31;
    const int khalf = lane >> 4;
    const int ln    = lane & 15;

    // B fragment bases (X̂ᵀ, 16-bit B 32x16 layout: lane = batch col,
    // K = kt*32 + khalf*8 + e (e<8), kt*32 + 16 + khalf*8 + (e-8) (e>=8)) — invariant over u.
    const _Float16* pb0 = &xt[ln * KPAD + khalf * 8];
    const _Float16* pb1 = pb0 + TILEH;

    for (int u = wave; u < UNITS; u += 8) {       // uniform per wave: EXEC all-1s
        const int o  = u / NTILES;
        const int nt = u - o * NTILES;
        const int n0 = nt * 16;

        // prefetch next unit's A fragments (gfx1250 global_prefetch path)
        if (u + 8 < UNITS)
            __builtin_prefetch(mf + ((size_t)((u + 8) * KTILES * 32 + lane) << 4), 0, 1);

        const _Float16* mfu = mf + ((size_t)(u * KTILES * 32 + lane) << 4);

        v8f acc0 = {}, acc1 = {};
#pragma unroll
        for (int kt = 0; kt < KTILES; ++kt) {
            v16h a, b0, b1;
            const _Float16* pa = mfu + (size_t)kt * 32 * 16;   // 32 contiguous bytes/lane
            ((uint4*)&a)[0] = ((const uint4*)pa)[0];
            ((uint4*)&a)[1] = ((const uint4*)pa)[1];

            ((uint4*)&b0)[0] = *(const uint4*)(pb0 + kt * 32);
            ((uint4*)&b0)[1] = *(const uint4*)(pb0 + kt * 32 + 16);
            ((uint4*)&b1)[0] = *(const uint4*)(pb1 + kt * 32);
            ((uint4*)&b1)[1] = *(const uint4*)(pb1 + kt * 32 + 16);

            acc0 = __builtin_amdgcn_wmma_f32_16x16x32_f16(
                false, a, false, b0, (short)0, acc0, false, false);
            acc1 = __builtin_amdgcn_wmma_f32_16x16x32_f16(
                false, a, false, b1, (short)0, acc1, false, false);
        }

        // D layout: lane ln = batch row; vgpr v = poly col n0 + v + 8*khalf.
        // s = sum_v acc[v] * x̂[row, n0+v+8*khalf]  (8 contiguous halves)
        {
            v8h xv;
            *(uint4*)&xv = *(const uint4*)&xt[ln * KPAD + n0 + khalf * 8];
            float s = 0.0f;
#pragma unroll
            for (int v = 0; v < 8; ++v) s = fmaf((float)xv[v], acc0[v], s);
            atomicAdd(&q[ln * NOUT + o], s);      // ds_add_f32
        }
        {
            v8h xv;
            *(uint4*)&xv = *(const uint4*)&xt[TILEH + ln * KPAD + n0 + khalf * 8];
            float s = 0.0f;
#pragma unroll
            for (int v = 0; v < 8; ++v) s = fmaf((float)xv[v], acc1[v], s);
            atomicAdd(&q[(16 + ln) * NOUT + o], s);
        }
    }
    __syncthreads();

    for (int i = tid; i < 32 * NOUT; i += 256) {
        const int row = i / NOUT, o = i - row * NOUT;
        out[(size_t)(bt * 32 + row) * NOUT + o] = q[i];
    }
}

// ---------------------------------------------------------------------------
extern "C" void kernel_launch(void* const* d_in, const int* in_sizes, int n_in,
                              void* d_out, int out_size, void* d_ws, size_t ws_size,
                              hipStream_t stream) {
    const float* x = (const float*)d_in[0];   // (4096, 256) f32
    const float* W = (const float*)d_in[1];   // (10, 33153) f32
    float* out = (float*)d_out;               // (4096, 10) f32

    _Float16* xh = (_Float16*)d_ws;                                     // 2,359,296 B
    _Float16* mf = (_Float16*)((char*)d_ws + (size_t)BATCH * KPAD * 2); // 1,566,720 B

    {
        int n = BATCH * KPAD;
        build_xhat<<<(n + 255) / 256, 256, 0, stream>>>(x, xh);
    }
    {
        int n = UNITS * KTILES * 32;
        build_mfrag<<<(n + 255) / 256, 256, 0, stream>>>(W, mf);
    }
    poly_wmma<<<BATCH / 32, 256, 0, stream>>>(xh, mf, out);
}
